// MoE_76459007803626
// MI455X (gfx1250) — matrix operand order repaired and checked by reading, not compiled, not run on hardware
//
#include <hip/hip_runtime.h>
#include <math.h>

// ---------- types ----------
typedef __bf16        v16bf  __attribute__((ext_vector_type(16)));
typedef __bf16        bf16x2 __attribute__((ext_vector_type(2)));
typedef float         v8f    __attribute__((ext_vector_type(8)));
typedef unsigned int  u32x4  __attribute__((ext_vector_type(4)));
typedef float         f32x4  __attribute__((ext_vector_type(4)));
typedef unsigned int  u32x2  __attribute__((ext_vector_type(2)));

#define LDS_ROW   40          // padded row stride in ushorts (32 data + 8 pad)
#define LDS_ROW_W 20          // same, in dwords

// fast fp32->bf16, round-half-up (one add + shift)
__device__ __forceinline__ unsigned short f2bf(float f) {
  return (unsigned short)((__float_as_uint(f) + 0x8000u) >> 16);
}
// pack two fp32 -> packed bf16x2 (lo = a, hi = b):
// 2x v_add + 1x v_perm_b32 (merge high halves)
__device__ __forceinline__ unsigned int pk_bf16(float a, float b) {
#if __has_builtin(__builtin_amdgcn_cvt_pk_bf16_f32)
  bf16x2 r = __builtin_amdgcn_cvt_pk_bf16_f32(a, b);   // v_cvt_pk_bf16_f32
  return __builtin_bit_cast(unsigned int, r);
#else
  unsigned int ua = __float_as_uint(a) + 0x8000u;
  unsigned int ub = __float_as_uint(b) + 0x8000u;
  // dest bytes {0,1} = src0(ua) bytes {2,3}; dest bytes {2,3} = src1(ub) bytes {2,3}
  return __builtin_amdgcn_perm(ua, ub, 0x03020706u);   // v_perm_b32
#endif
}
__device__ __forceinline__ float bf2f(unsigned short h) {
  return __uint_as_float(((unsigned int)h) << 16);
}

struct Q2 { u32x4 a, b; };
__device__ __forceinline__ v16bf load_frag(const unsigned short* p0,
                                           const unsigned short* p1) {
  Q2 s;
  s.a = *(const u32x4*)p0;
  s.b = *(const u32x4*)p1;
  return __builtin_bit_cast(v16bf, s);
}

// ---------- CDNA5 async global->LDS DMA (ASYNCcnt-tracked) ----------
__device__ __forceinline__ void async_ld_b128(unsigned ldsOff, const void* gaddr) {
  asm volatile("global_load_async_to_lds_b128 %0, %1, off"
               :: "v"(ldsOff), "v"(gaddr)
               : "memory");
}
__device__ __forceinline__ void wait_async0() {
#if __has_builtin(__builtin_amdgcn_s_wait_asynccnt)
  __builtin_amdgcn_s_wait_asynccnt(0);
#else
  asm volatile("s_wait_asynccnt 0" ::: "memory");
#endif
}
__device__ __forceinline__ unsigned lds_off(const void* p) {
  return (unsigned)(size_t)p;   // flat LDS addr: low 32 bits are the LDS offset
}

// ---------- 1. router ----------
__global__ void moe_router(const float* __restrict__ x, const float* __restrict__ gw,
                           int T, int H,
                           float* __restrict__ wslot, int* __restrict__ slot_e,
                           int* __restrict__ posn, int* __restrict__ counts) {
  const int lane = threadIdx.x & 31;
  const int wv   = threadIdx.x >> 5;
  const int t    = blockIdx.x * 8 + wv;
  if (t >= T) return;

  float acc[8];
#pragma unroll
  for (int e = 0; e < 8; ++e) acc[e] = 0.f;
  const float* xr = x + (size_t)t * H;
  for (int h = lane; h < H; h += 32) {
    float xv = xr[h];
#pragma unroll
    for (int e = 0; e < 8; ++e) acc[e] += xv * gw[e * H + h];
  }
#pragma unroll
  for (int e = 0; e < 8; ++e) {
#pragma unroll
    for (int off = 16; off > 0; off >>= 1)
      acc[e] += __shfl_xor(acc[e], off, 32);
  }
  if (lane == 0) {
    float m = acc[0];
#pragma unroll
    for (int e = 1; e < 8; ++e) m = fmaxf(m, acc[e]);
    float p[8], s = 0.f;
#pragma unroll
    for (int e = 0; e < 8; ++e) { p[e] = __expf(acc[e] - m); s += p[e]; }
    float inv_s = 1.f / s;
#pragma unroll
    for (int e = 0; e < 8; ++e) p[e] *= inv_s;
    int i0 = 0;
#pragma unroll
    for (int e = 1; e < 8; ++e) if (p[e] > p[i0]) i0 = e;
    int i1 = (i0 == 0) ? 1 : 0;
#pragma unroll
    for (int e = 0; e < 8; ++e) if (e != i0 && p[e] > p[i1]) i1 = e;
    float w0 = p[i0], w1 = p[i1];
    float inv = 1.f / (w0 + w1);
    w0 *= inv; w1 *= inv;
    int q0 = atomicAdd(&counts[i0], 1);
    int q1 = atomicAdd(&counts[i1], 1);
    wslot[t * 2 + 0] = w0;  wslot[t * 2 + 1] = w1;
    slot_e[t * 2 + 0] = i0; slot_e[t * 2 + 1] = i1;
    posn[t * 2 + 0] = q0;   posn[t * 2 + 1] = q1;
  }
}

// ---------- 2. offsets ----------
__global__ void moe_offsets(const int* __restrict__ counts, int* __restrict__ offs, int E) {
  if (threadIdx.x == 0 && blockIdx.x == 0) {
    int s = 0;
    for (int e = 0; e < E; ++e) { offs[e] = s; s += counts[e]; }
    offs[E] = s;
  }
}

// ---------- 3. scatter ----------
__global__ void moe_scatter(const int* __restrict__ slot_e, const int* __restrict__ posn,
                            const float* __restrict__ wslot, const int* __restrict__ offs,
                            int* __restrict__ tok_list, int* __restrict__ pidx,
                            float* __restrict__ wpair, int T) {
  int t = blockIdx.x * blockDim.x + threadIdx.x;
  if (t >= T) return;
#pragma unroll
  for (int s = 0; s < 2; ++s) {
    int e = slot_e[t * 2 + s];
    int p = offs[e] + posn[t * 2 + s];
    tok_list[p] = t;
    wpair[p]    = wslot[t * 2 + s];
    pidx[t * 2 + s] = p;
  }
}

// ---------- 4. x fp32 -> bf16 ----------
__global__ void moe_xconv(const float* __restrict__ x, unsigned short* __restrict__ xb,
                          size_t n4) {
  size_t i = (size_t)blockIdx.x * blockDim.x + threadIdx.x;
  if (i >= n4) return;
  f32x4 v = *(const f32x4*)(x + i * 4);
  u32x2 o;
  o[0] = pk_bf16(v[0], v[1]);
  o[1] = pk_bf16(v[2], v[3]);
  *(u32x2*)(xb + i * 4) = o;
}

// ---------- 5. grouped GEMM1: h = silu(xg@w1) * (xg@w3) ----------
__global__ __launch_bounds__(256)
void moe_gemm1(const unsigned short* __restrict__ xb,
               const float* __restrict__ w1, const float* __restrict__ w3,
               const int* __restrict__ counts, const int* __restrict__ offs,
               const int* __restrict__ tok_list,
               unsigned short* __restrict__ hbuf,
               int H, int F, int tilesMax) {
  __shared__ unsigned short lA [2][128 * LDS_ROW];
  __shared__ unsigned short lB1[2][128 * LDS_ROW];
  __shared__ unsigned short lB3[2][128 * LDS_ROW];

  const int e    = blockIdx.x / tilesMax;
  const int tile = blockIdx.x % tilesMax;
  const int cnt  = counts[e];
  if (tile * 128 >= cnt) return;
  const int off  = offs[e];
  const int f0   = blockIdx.y * 128;

  const int tid  = threadIdx.x;
  const int lane = tid & 31, wv = tid >> 5;
  const int mw = (wv & 1) * 64;
  const int nw = (wv >> 1) * 32;

  // A staging via async DMA: 2 threads per row, 32 bytes each
  const int arow  = tid >> 1;
  const int ahalf = (tid & 1) * 16;
  const int tk_i  = tile * 128 + arow;
  const int tok   = (tk_i < cnt) ? tok_list[off + tk_i] : tok_list[off];
  const unsigned short* aSrc = xb + (size_t)tok * H + ahalf;
  const unsigned aLds0   = lds_off(&lA[0][arow * LDS_ROW + ahalf]);
  const unsigned aLdsBuf = (unsigned)sizeof(lA[0]);

  // B staging: thread owns k-pair (2*kp, 2*kp+1) x 8 columns
  const int kp = tid & 15;
  const int nn = (tid >> 4) * 8;
  const size_t wBase = (size_t)e * H * F + (size_t)(2 * kp) * F + f0 + nn;
  const float* b1Src = w1 + wBase;
  const float* b3Src = w3 + wBase;

  const int steps = H / 32;

  // prologue: async A for step 0, global B for step 0
  async_ld_b128(aLds0,      aSrc);
  async_ld_b128(aLds0 + 16, aSrc + 8);
  f32x4 rb1[4], rb3[4];
#pragma unroll
  for (int q = 0; q < 2; ++q) {
    rb1[q]     = *(const f32x4*)(b1Src + q * 4);
    rb1[q + 2] = *(const f32x4*)(b1Src + F + q * 4);
    rb3[q]     = *(const f32x4*)(b3Src + q * 4);
    rb3[q + 2] = *(const f32x4*)(b3Src + F + q * 4);
  }

  v8f acc1[4][2], acc3[4][2];
#pragma unroll
  for (int i = 0; i < 4; ++i)
#pragma unroll
    for (int j = 0; j < 2; ++j)
#pragma unroll
      for (int r = 0; r < 8; ++r) { acc1[i][j][r] = 0.f; acc3[i][j][r] = 0.f; }

  for (int s = 0; s < steps; ++s) {
    const int buf = s & 1;
    // commit staged B registers to LDS as packed bf16x2 dword stores
    {
      unsigned int* w1d = (unsigned int*)&lB1[buf][0];
      unsigned int* w3d = (unsigned int*)&lB3[buf][0];
      const float* f1 = (const float*)rb1;
      const float* f3 = (const float*)rb3;
#pragma unroll
      for (int q = 0; q < 8; ++q) {
        w1d[(nn + q) * LDS_ROW_W + kp] = pk_bf16(f1[q], f1[q + 8]);
        w3d[(nn + q) * LDS_ROW_W + kp] = pk_bf16(f3[q], f3[q + 8]);
      }
    }
    wait_async0();            // this step's A DMA has landed
    __syncthreads();
    // issue next step's async A + global B (overlaps WMMA below)
    if (s + 1 < steps) {
      const int k0 = (s + 1) * 32;
      const unsigned d = aLds0 + (unsigned)((s + 1) & 1) * aLdsBuf;
      async_ld_b128(d,      aSrc + k0);
      async_ld_b128(d + 16, aSrc + k0 + 8);
      const size_t o = (size_t)k0 * F;
#pragma unroll
      for (int q = 0; q < 2; ++q) {
        rb1[q]     = *(const f32x4*)(b1Src + o + q * 4);
        rb1[q + 2] = *(const f32x4*)(b1Src + o + F + q * 4);
        rb3[q]     = *(const f32x4*)(b3Src + o + q * 4);
        rb3[q + 2] = *(const f32x4*)(b3Src + o + F + q * 4);
      }
    }
    // compute: 16 WMMAs per wave per K-step
    const int akk = (lane < 16) ? 0 : 8;
    const int bkk = (lane < 16) ? 0 : 8;
    v16bf af[4];
#pragma unroll
    for (int i = 0; i < 4; ++i) {
      const unsigned short* pa = &lA[buf][(mw + i * 16 + (lane & 15)) * LDS_ROW + akk];
      af[i] = load_frag(pa, pa + 16);
    }
#pragma unroll
    for (int j = 0; j < 2; ++j) {
      const unsigned short* pb1 = &lB1[buf][(nw + j * 16 + (lane & 15)) * LDS_ROW + bkk];
      const unsigned short* pb3 = &lB3[buf][(nw + j * 16 + (lane & 15)) * LDS_ROW + bkk];
      v16bf b1f = load_frag(pb1, pb1 + 16);
      v16bf b3f = load_frag(pb3, pb3 + 16);
#pragma unroll
      for (int i = 0; i < 4; ++i) {
        acc1[i][j] = __builtin_amdgcn_wmma_f32_16x16x32_bf16(
            false, af[i], false, b1f, (short)0, acc1[i][j], false, false);
        acc3[i][j] = __builtin_amdgcn_wmma_f32_16x16x32_bf16(
            false, af[i], false, b3f, (short)0, acc3[i][j], false, false);
      }
    }
  }

  // epilogue: silu(h1)*h3 -> bf16 hbuf
  const int mbase = mw + ((lane >> 4) << 3);
  const int nbase = nw + (lane & 15);
#pragma unroll
  for (int i = 0; i < 4; ++i)
#pragma unroll
    for (int j = 0; j < 2; ++j)
#pragma unroll
      for (int r = 0; r < 8; ++r) {
        const int mrow = mbase + i * 16 + r;
        if (tile * 128 + mrow < cnt) {
          float h1 = acc1[i][j][r];
          float h3 = acc3[i][j][r];
          float v  = h1 / (1.f + __expf(-h1)) * h3;
          size_t p = (size_t)(off + tile * 128 + mrow);
          hbuf[p * F + f0 + nbase + j * 16] = f2bf(v);
        }
      }
}

// ---------- 6. grouped GEMM2: opair = wpair * (h @ w2) ----------
__global__ __launch_bounds__(256)
void moe_gemm2(const unsigned short* __restrict__ hbuf,
               const float* __restrict__ w2,
               const int* __restrict__ counts, const int* __restrict__ offs,
               const float* __restrict__ wpair,
               unsigned short* __restrict__ opair,
               int F, int H, int tilesMax) {
  __shared__ unsigned short lA[2][128 * LDS_ROW];
  __shared__ unsigned short lB[2][128 * LDS_ROW];

  const int e    = blockIdx.x / tilesMax;
  const int tile = blockIdx.x % tilesMax;
  const int cnt  = counts[e];
  if (tile * 128 >= cnt) return;
  const int off  = offs[e];
  const int n0   = blockIdx.y * 128;

  const int tid  = threadIdx.x;
  const int lane = tid & 31, wv = tid >> 5;
  const int mw = (wv & 1) * 64;
  const int nw = (wv >> 1) * 32;

  const int arow  = tid >> 1;
  const int ahalf = (tid & 1) * 16;
  const unsigned short* aSrc =
      hbuf + (size_t)(off + tile * 128 + arow) * F + ahalf;
  const unsigned aLds0   = lds_off(&lA[0][arow * LDS_ROW + ahalf]);
  const unsigned aLdsBuf = (unsigned)sizeof(lA[0]);

  const int kp = tid & 15;
  const int nn = (tid >> 4) * 8;
  const float* bSrc = w2 + (size_t)e * F * H + (size_t)(2 * kp) * H + n0 + nn;

  const int steps = F / 32;

  async_ld_b128(aLds0,      aSrc);
  async_ld_b128(aLds0 + 16, aSrc + 8);
  f32x4 rb[4];
#pragma unroll
  for (int q = 0; q < 2; ++q) {
    rb[q]     = *(const f32x4*)(bSrc + q * 4);
    rb[q + 2] = *(const f32x4*)(bSrc + H + q * 4);
  }

  v8f acc[4][2];
#pragma unroll
  for (int i = 0; i < 4; ++i)
#pragma unroll
    for (int j = 0; j < 2; ++j)
#pragma unroll
      for (int r = 0; r < 8; ++r) acc[i][j][r] = 0.f;

  for (int s = 0; s < steps; ++s) {
    const int buf = s & 1;
    {
      unsigned int* wd = (unsigned int*)&lB[buf][0];
      const float* fb = (const float*)rb;
#pragma unroll
      for (int q = 0; q < 8; ++q)
        wd[(nn + q) * LDS_ROW_W + kp] = pk_bf16(fb[q], fb[q + 8]);
    }
    wait_async0();
    __syncthreads();
    if (s + 1 < steps) {
      const int k0 = (s + 1) * 32;
      const unsigned d = aLds0 + (unsigned)((s + 1) & 1) * aLdsBuf;
      async_ld_b128(d,      aSrc + k0);
      async_ld_b128(d + 16, aSrc + k0 + 8);
      const size_t o = (size_t)k0 * H;
#pragma unroll
      for (int q = 0; q < 2; ++q) {
        rb[q]     = *(const f32x4*)(bSrc + o + q * 4);
        rb[q + 2] = *(const f32x4*)(bSrc + o + H + q * 4);
      }
    }
    const int akk = (lane < 16) ? 0 : 8;
    const int bkk = (lane < 16) ? 0 : 8;
    v16bf af[4];
#pragma unroll
    for (int i = 0; i < 4; ++i) {
      const unsigned short* pa = &lA[buf][(mw + i * 16 + (lane & 15)) * LDS_ROW + akk];
      af[i] = load_frag(pa, pa + 16);
    }
#pragma unroll
    for (int j = 0; j < 2; ++j) {
      const unsigned short* pb = &lB[buf][(nw + j * 16 + (lane & 15)) * LDS_ROW + bkk];
      v16bf bf = load_frag(pb, pb + 16);
#pragma unroll
      for (int i = 0; i < 4; ++i)
        acc[i][j] = __builtin_amdgcn_wmma_f32_16x16x32_bf16(
            false, af[i], false, bf, (short)0, acc[i][j], false, false);
    }
  }

  const int mbase = mw + ((lane >> 4) << 3);
  const int nbase = nw + (lane & 15);
#pragma unroll
  for (int i = 0; i < 4; ++i)
#pragma unroll
    for (int j = 0; j < 2; ++j)
#pragma unroll
      for (int r = 0; r < 8; ++r) {
        const int mrow = mbase + i * 16 + r;
        if (tile * 128 + mrow < cnt) {
          size_t p = (size_t)(off + tile * 128 + mrow);
          float v = wpair[p] * acc[i][j][r];
          opair[p * H + n0 + nbase + j * 16] = f2bf(v);
        }
      }
}

// ---------- 7. combine ----------
__global__ void moe_combine(const unsigned short* __restrict__ opair,
                            const int* __restrict__ pidx,
                            float* __restrict__ out, int T, int H) {
  size_t i = (size_t)blockIdx.x * blockDim.x + threadIdx.x;
  size_t total = (size_t)T * (H / 8);
  if (i >= total) return;
  const int t = (int)(i / (H / 8));
  const int c = (int)(i % (H / 8)) * 8;
  const int p0 = pidx[t * 2 + 0];
  const int p1 = pidx[t * 2 + 1];
  u32x4 A = *(const u32x4*)(opair + (size_t)p0 * H + c);
  u32x4 B = *(const u32x4*)(opair + (size_t)p1 * H + c);
  f32x4 o0, o1;
#pragma unroll
  for (int k = 0; k < 4; ++k) {
    float a0 = bf2f((unsigned short)(A[k] & 0xFFFF));
    float a1 = bf2f((unsigned short)(A[k] >> 16));
    float b0 = bf2f((unsigned short)(B[k] & 0xFFFF));
    float b1 = bf2f((unsigned short)(B[k] >> 16));
    if (k < 2) { o0[2 * k] = a0 + b0; o0[2 * k + 1] = a1 + b1; }
    else       { o1[2 * (k - 2)] = a0 + b0; o1[2 * (k - 2) + 1] = a1 + b1; }
  }
  float* dst = out + (size_t)t * H + c;
  *(f32x4*)(dst + 0) = o0;
  *(f32x4*)(dst + 4) = o1;
}

// ---------- host launcher ----------
extern "C" void kernel_launch(void* const* d_in, const int* in_sizes, int n_in,
                              void* d_out, int out_size, void* d_ws, size_t ws_size,
                              hipStream_t stream) {
  const float* x  = (const float*)d_in[0];
  const float* gw = (const float*)d_in[1];
  const float* w1 = (const float*)d_in[2];
  const float* w2 = (const float*)d_in[3];
  const float* w3 = (const float*)d_in[4];

  const int E = 8;
  const int H = in_sizes[1] / E;
  const int F = in_sizes[2] / (E * H);
  const int T = in_sizes[0] / H;
  const int P = T * 2;

  char* ws = (char*)d_ws;
  size_t o = 0;
  auto alloc = [&](size_t bytes) -> void* {
    void* p = ws + o;
    o = (o + bytes + 255) & ~(size_t)255;
    return p;
  };
  unsigned short* xb    = (unsigned short*)alloc((size_t)T * H * 2);
  unsigned short* hbuf  = (unsigned short*)alloc((size_t)P * F * 2);
  unsigned short* opair = (unsigned short*)alloc((size_t)P * H * 2);
  float* wslot   = (float*)alloc((size_t)T * 2 * 4);
  int*   slot_e  = (int*)  alloc((size_t)T * 2 * 4);
  int*   posn    = (int*)  alloc((size_t)T * 2 * 4);
  int*   pidx    = (int*)  alloc((size_t)T * 2 * 4);
  float* wpair   = (float*)alloc((size_t)P * 4);
  int*   tok_list= (int*)  alloc((size_t)P * 4);
  int*   counts  = (int*)  alloc((size_t)E * 4);
  int*   offs    = (int*)  alloc((size_t)(E + 1) * 4);
  (void)ws_size; (void)n_in; (void)out_size;

  hipMemsetAsync(counts, 0, E * sizeof(int), stream);

  moe_router<<<(T + 7) / 8, 256, 0, stream>>>(x, gw, T, H, wslot, slot_e, posn, counts);
  moe_offsets<<<1, 1, 0, stream>>>(counts, offs, E);
  moe_scatter<<<(T + 255) / 256, 256, 0, stream>>>(slot_e, posn, wslot, offs,
                                                   tok_list, pidx, wpair, T);
  {
    size_t n4 = (size_t)T * H / 4;
    moe_xconv<<<(unsigned)((n4 + 255) / 256), 256, 0, stream>>>(x, xb, n4);
  }
  const int tilesMax = (T + 127) / 128;
  moe_gemm1<<<dim3(E * tilesMax, F / 128), 256, 0, stream>>>(
      xb, w1, w3, counts, offs, tok_list, hbuf, H, F, tilesMax);
  moe_gemm2<<<dim3(E * tilesMax, H / 128), 256, 0, stream>>>(
      hbuf, w2, counts, offs, wpair, opair, F, H, tilesMax);
  {
    size_t n = (size_t)T * (H / 8);
    moe_combine<<<(unsigned)((n + 255) / 256), 256, 0, stream>>>(opair, pidx,
                                                                 (float*)d_out, T, H);
  }
}
